// vgae_64304250356120
// MI455X (gfx1250) — compile-verified, matrix operand order checked by reading, not run on hardware
//
#include <hip/hip_runtime.h>
#include <hip/hip_bf16.h>

// CDNA5 (gfx1250) wave32 WMMA types
typedef __attribute__((ext_vector_type(16))) __bf16 v16bf;
typedef __attribute__((ext_vector_type(8)))  __bf16 v8bf;
typedef __attribute__((ext_vector_type(8)))  float  v8f;

static __device__ __forceinline__ __bf16 tobf(float f) { return (__bf16)f; }

// ---------------------------------------------------------------------------
// Kernel 0: zero the scalar output (graph-capture safe, no hipMemset)
// ---------------------------------------------------------------------------
__global__ void vgae_init_kernel(float* out) {
    if (threadIdx.x == 0) out[0] = 0.0f;
}

// ---------------------------------------------------------------------------
// Kernel 1: KL term.  sum over N*D of (1 + 2*log(sd+1e-6) - mu^2 - sd^2),
// scaled by -0.5/N, atomically accumulated into out[0].
// ---------------------------------------------------------------------------
__global__ __launch_bounds__(256) void vgae_kl_kernel(
    const float* __restrict__ xm, const float* __restrict__ xsd,
    float* __restrict__ out, int nd, float scale)
{
    float p = 0.0f;
    int stride = gridDim.x * blockDim.x;
    for (int i = blockIdx.x * blockDim.x + threadIdx.x; i < nd; i += stride) {
        float mu = xm[i], sd = xsd[i];
        p += 1.0f + 2.0f * __logf(sd + 1e-6f) - mu * mu - sd * sd;
    }
    __shared__ float red[256];
    red[threadIdx.x] = p;
    __syncthreads();
    for (int off = 128; off > 0; off >>= 1) {
        if (threadIdx.x < (unsigned)off) red[threadIdx.x] += red[threadIdx.x + off];
        __syncthreads();
    }
    if (threadIdx.x == 0) unsafeAtomicAdd(out, red[0] * scale);
}

// ---------------------------------------------------------------------------
// Kernel 2: edge loss via bf16 WMMA.
// 256 threads = 8 waves; each wave handles one tile of 16 edges.
//   A (16x128)  = relu(x[src] * x[dst]) per edge row, bf16
//   H (16x128)  = relu(A @ W1 + b1)   -> 8 N-tiles x 4 K-steps of
//                 v_wmma_f32_16x16x32_bf16
//   logit (16)  = H @ W2 + b2  (epilogue: per-lane partial + 16-lane butterfly)
//   loss       += -log(clip(sigmoid(logit), 1e-12, 1)) = clamped softplus(-z)
// ---------------------------------------------------------------------------
__global__ __launch_bounds__(256) void vgae_edge_kernel(
    const float* __restrict__ x,  const float* __restrict__ W1,
    const float* __restrict__ b1, const float* __restrict__ W2,
    const float* __restrict__ b2, const int*  __restrict__ ei,
    float* __restrict__ out, int E, int tiles)
{
    __shared__ __bf16 w1t[128 * 128];   // W1 transposed: w1t[n*128 + k], 32 KB
    __shared__ float  b1s[128];
    __shared__ float  w2s[128];
    __shared__ float  red[256];

    const int tid = threadIdx.x;

    // Stage W1 (transposed, bf16) + b1 + W2 into LDS once per block.
    for (int idx = tid; idx < 128 * 128; idx += 256) {
        int k = idx >> 7, n = idx & 127;           // W1 is [k][n] row-major
        w1t[(n << 7) + k] = tobf(W1[idx]);
    }
    if (tid < 128) { b1s[tid] = b1[tid]; w2s[tid] = W2[tid]; }
    __syncthreads();

    const int lane = tid & 31;
    const int wave = tid >> 5;
    const int tile = blockIdx.x * 8 + wave;

    float partial = 0.0f;

    if (tile < tiles) {
        const int m  = lane & 15;            // A-row handled by this lane
        const int kb = (lane >> 4) << 3;     // K sub-offset: 0 (lo half) / 8 (hi half)
        const int e  = tile * 16 + m;
        const int sI = (e < E) ? ei[e]     : 0;
        const int tI = (e < E) ? ei[E + e] : 0;
        const float* xs_ = x + (long)sI * 128;
        const float* xt_ = x + (long)tI * 128;

        // Build 4 A fragments (K = 0..127 in 32-wide steps).
        // Element i of the fragment holds K = kk*32 + kb + (i<8 ? i : i+8).
        v16bf afr[4];
#pragma unroll
        for (int kk = 0; kk < 4; ++kk) {
            const int Bo = kk * 32 + kb;
            float4 s0 = *(const float4*)(xs_ + Bo);
            float4 s1 = *(const float4*)(xs_ + Bo + 4);
            float4 s2 = *(const float4*)(xs_ + Bo + 16);
            float4 s3 = *(const float4*)(xs_ + Bo + 20);
            float4 t0 = *(const float4*)(xt_ + Bo);
            float4 t1 = *(const float4*)(xt_ + Bo + 4);
            float4 t2 = *(const float4*)(xt_ + Bo + 16);
            float4 t3 = *(const float4*)(xt_ + Bo + 20);
            v16bf a;
            a[0]  = tobf(fmaxf(s0.x * t0.x, 0.0f));
            a[1]  = tobf(fmaxf(s0.y * t0.y, 0.0f));
            a[2]  = tobf(fmaxf(s0.z * t0.z, 0.0f));
            a[3]  = tobf(fmaxf(s0.w * t0.w, 0.0f));
            a[4]  = tobf(fmaxf(s1.x * t1.x, 0.0f));
            a[5]  = tobf(fmaxf(s1.y * t1.y, 0.0f));
            a[6]  = tobf(fmaxf(s1.z * t1.z, 0.0f));
            a[7]  = tobf(fmaxf(s1.w * t1.w, 0.0f));
            a[8]  = tobf(fmaxf(s2.x * t2.x, 0.0f));
            a[9]  = tobf(fmaxf(s2.y * t2.y, 0.0f));
            a[10] = tobf(fmaxf(s2.z * t2.z, 0.0f));
            a[11] = tobf(fmaxf(s2.w * t2.w, 0.0f));
            a[12] = tobf(fmaxf(s3.x * t3.x, 0.0f));
            a[13] = tobf(fmaxf(s3.y * t3.y, 0.0f));
            a[14] = tobf(fmaxf(s3.z * t3.z, 0.0f));
            a[15] = tobf(fmaxf(s3.w * t3.w, 0.0f));
            afr[kk] = a;
        }

        const int nl = lane & 15;
        float plog[8];
#pragma unroll
        for (int r = 0; r < 8; ++r) plog[r] = 0.0f;

#pragma unroll
        for (int nt = 0; nt < 8; ++nt) {
            const int n  = nt * 16 + nl;          // output column for this lane
            const float bv = b1s[n];
            v8f acc = {bv, bv, bv, bv, bv, bv, bv, bv};   // C init = b1 broadcast
#pragma unroll
            for (int kk = 0; kk < 4; ++kk) {
                const int Bo  = kk * 32 + kb;
                const int off = (n << 7) + Bo;     // contiguous K run in w1t
                v8bf lo = *(const v8bf*)&w1t[off];        // K = Bo..Bo+7
                v8bf hi = *(const v8bf*)&w1t[off + 16];   // K = Bo+16..Bo+23
                v16bf bfr;
#pragma unroll
                for (int i = 0; i < 8; ++i) { bfr[i] = lo[i]; bfr[i + 8] = hi[i]; }
                acc = __builtin_amdgcn_wmma_f32_16x16x32_bf16(
                    /*neg_a=*/false, afr[kk], /*neg_b=*/false, bfr,
                    /*c_mod=*/(short)0, acc, /*reuse_a=*/false, /*reuse_b=*/false);
            }
            // Epilogue: relu(H) * W2[n], accumulated per row-register.
            const float w2v = w2s[n];
#pragma unroll
            for (int r = 0; r < 8; ++r)
                plog[r] += fmaxf(acc[r], 0.0f) * w2v;
        }

        // Butterfly-sum across the 16 lanes of each half (masks 1,2,4,8 stay
        // inside a half).  Low half holds rows 0..7, high half rows 8..15.
#pragma unroll
        for (int r = 0; r < 8; ++r) {
#pragma unroll
            for (int msk = 1; msk < 16; msk <<= 1)
                plog[r] += __shfl_xor(plog[r], msk, 32);
        }

        if (nl == 0) {
            const float b2v = b2[0];
            const int rowBase = tile * 16 + ((lane >> 4) << 3);
#pragma unroll
            for (int r = 0; r < 8; ++r) {
                if (rowBase + r < E) {
                    float z = plog[r] + b2v;
                    // -log(clip(sigmoid(z),1e-12,1)); -log(1e-12)=27.6310211159
                    float l = (z > -27.6310211159f) ? log1pf(__expf(-z))
                                                    : 27.6310211159f;
                    partial += l;
                }
            }
        }
    }

    red[tid] = partial;
    __syncthreads();
    for (int off = 128; off > 0; off >>= 1) {
        if (tid < (unsigned)off) red[tid] += red[tid + off];
        __syncthreads();
    }
    if (tid == 0) unsafeAtomicAdd(out, red[0] * (1.0f / (float)E));
}

// ---------------------------------------------------------------------------
extern "C" void kernel_launch(void* const* d_in, const int* in_sizes, int n_in,
                              void* d_out, int out_size, void* d_ws, size_t ws_size,
                              hipStream_t stream) {
    const float* x   = (const float*)d_in[0];
    const float* xm  = (const float*)d_in[1];
    const float* xsd = (const float*)d_in[2];
    const float* W1  = (const float*)d_in[3];
    const float* b1  = (const float*)d_in[4];
    const float* W2  = (const float*)d_in[5];
    const float* b2  = (const float*)d_in[6];
    const int*   ei  = (const int*)d_in[7];
    // d_in[8] (edge_index_batch) is all-zero with B==1: pooling cancels exactly.

    const int ND = in_sizes[0];        // N * 128
    const int N  = ND >> 7;
    const int E  = in_sizes[7] / 2;
    float* out = (float*)d_out;

    vgae_init_kernel<<<1, 64, 0, stream>>>(out);

    int klBlocks = (ND + 256 * 16 - 1) / (256 * 16);
    if (klBlocks < 1)    klBlocks = 1;
    if (klBlocks > 4096) klBlocks = 4096;
    vgae_kl_kernel<<<klBlocks, 256, 0, stream>>>(xm, xsd, out, ND, -0.5f / (float)N);

    const int tiles  = (E + 15) / 16;
    const int blocks = (tiles + 7) / 8;   // 8 waves/block, 1 tile/wave
    vgae_edge_kernel<<<blocks, 256, 0, stream>>>(x, W1, b1, W2, b2, ei, out, E, tiles);
}
